// GCN_72645076845136
// MI455X (gfx1250) — matrix operand order repaired and checked by reading, not compiled, not run on hardware
//
#include <hip/hip_runtime.h>
#include <hip/hip_bf16.h>

typedef __attribute__((ext_vector_type(16))) __bf16 v16bf;
typedef __attribute__((ext_vector_type(8)))  float  v8f;
typedef __attribute__((ext_vector_type(4)))  int    v4i;

#define NFEAT  512
#define NHID   256
#define NCLASS 64
#define KSTEPS (NFEAT / 32)      // 16 K-steps for GEMM1
#define BROW   40                // LDS row stride in bf16 elems (80B: 16B-aligned, depads banks)

#if __has_builtin(__builtin_amdgcn_global_load_async_to_lds_b128) && \
    __has_builtin(__builtin_amdgcn_s_wait_asynccnt)
#define HAVE_ASYNC_LDS 1
typedef __attribute__((address_space(1))) v4i gv4i;   // global 128-bit chunk
typedef __attribute__((address_space(3))) v4i lv4i;   // LDS 128-bit chunk
__device__ __forceinline__ void async_cp16(const void* g, void* l) {
  __builtin_amdgcn_global_load_async_to_lds_b128((gv4i*)g, (lv4i*)l, 0, 0);
}
#else
#define HAVE_ASYNC_LDS 0
#endif

__device__ __forceinline__ unsigned bf16r(float f) {
  unsigned u = __float_as_uint(f);
  return (u + 0x7FFFu + ((u >> 16) & 1u)) >> 16;     // round-to-nearest-even bf16
}
__device__ __forceinline__ unsigned pk2(float lo, float hi) {
  return bf16r(lo) | (bf16r(hi) << 16);
}

union FragU { unsigned u[8]; uint4 q[2]; v16bf v; };

// ---- W[K][N] fp32 -> WT[N][K] bf16 (transpose + convert) -------------------
__global__ void k_cvt_wT(const float* __restrict__ W, unsigned short* __restrict__ WT,
                         int K, int N) {
  int i = blockIdx.x * blockDim.x + threadIdx.x;
  if (i >= K * N) return;
  int k = i / N, n = i - k * N;
  WT[(size_t)n * K + k] = (unsigned short)bf16r(W[i]);
}

// ---- GEMM1: Y[nrows x 256] = X[nrows x 512](fp32->bf16) @ W1 (via W1T bf16)
// B tile (128 cols x 32 K) staged in LDS, double-buffered; next tile fetched
// with async global->LDS copies (ASYNCcnt) while WMMAs consume current tile.
__global__ __launch_bounds__(256) void k_gemm_x_w1(
    const float* __restrict__ X, const unsigned short* __restrict__ W1T,
    float* __restrict__ Y, int nrows) {
  __shared__ unsigned short Bs[2][128 * BROW];       // 2 x 10 KB

  const int tid   = threadIdx.x;
  const int lane  = tid & 31;
  const int wave  = tid >> 5;
  const int half  = lane >> 4;
  const int lm    = lane & 15;
  const int m0    = blockIdx.x * 128 + wave * 16;
  const int nbase = blockIdx.y * 128;

  int arow = m0 + lm;
  if (arow >= nrows) arow = nrows - 1;               // clamp tail loads; stores guarded
  const float* xrow = X + (size_t)arow * NFEAT;

  // cooperative B tile fetch: thread covers (row=tid/4, row+64), 16B segment tid%4
  const int brow = tid >> 2, bseg = tid & 3;
  const unsigned short* bsrc0 = W1T + (size_t)(nbase + brow) * NFEAT + bseg * 8;
  const unsigned short* bsrc1 = bsrc0 + (size_t)64 * NFEAT;
  const int bdst0 = brow * BROW + bseg * 8;
  const int bdst1 = (brow + 64) * BROW + bseg * 8;

  v8f acc[8];
  #pragma unroll
  for (int j = 0; j < 8; ++j) acc[j] = (v8f){0,0,0,0,0,0,0,0};

  // preload tile 0
#if HAVE_ASYNC_LDS
  async_cp16(bsrc0, &Bs[0][bdst0]);
  async_cp16(bsrc1, &Bs[0][bdst1]);
  __builtin_amdgcn_s_wait_asynccnt(0);
  __syncthreads();
#else
  *(uint4*)&Bs[0][bdst0] = *(const uint4*)bsrc0;
  *(uint4*)&Bs[0][bdst1] = *(const uint4*)bsrc1;
  __syncthreads();
#endif

  #pragma unroll 2
  for (int step = 0; step < KSTEPS; ++step) {
    const int k0  = step * 32;
    const int cur = step & 1;

#if HAVE_ASYNC_LDS
    if (step + 1 < KSTEPS) {                         // prefetch next tile into other buffer
      async_cp16(bsrc0 + k0 + 32, &Bs[cur ^ 1][bdst0]);
      async_cp16(bsrc1 + k0 + 32, &Bs[cur ^ 1][bdst1]);
    }
#else
    uint4 nx0, nx1;
    if (step + 1 < KSTEPS) {
      nx0 = *(const uint4*)(bsrc0 + k0 + 32);
      nx1 = *(const uint4*)(bsrc1 + k0 + 32);
    }
#endif

    // A fragment (16x32 bf16) from fp32, per ISA 16-bit A layout:
    // lanes 0-15 K=[k0,k0+8)+[k0+16,k0+24); lanes 16-31 the +8 chunks.
    const int kA = k0 + half * 8;
    float4 c0 = *(const float4*)(xrow + kA);
    float4 c1 = *(const float4*)(xrow + kA + 4);
    float4 c2 = *(const float4*)(xrow + kA + 16);
    float4 c3 = *(const float4*)(xrow + kA + 20);
    FragU a;
    a.u[0]=pk2(c0.x,c0.y); a.u[1]=pk2(c0.z,c0.w);
    a.u[2]=pk2(c1.x,c1.y); a.u[3]=pk2(c1.z,c1.w);
    a.u[4]=pk2(c2.x,c2.y); a.u[5]=pk2(c2.z,c2.w);
    a.u[6]=pk2(c3.x,c3.y); a.u[7]=pk2(c3.z,c3.w);
    __builtin_prefetch(xrow + k0 + 32, 0, 1);        // global_prefetch_b8 next A chunk

    #pragma unroll
    for (int j = 0; j < 8; ++j) {
      // B fragment from LDS: lane = N column, K = [8h,8h+8)+[16+8h,24+8h) per half
      const unsigned short* bp = &Bs[cur][(j * 16 + lm) * BROW + half * 8];
      FragU b;
      b.q[0] = ((const uint4*)bp)[0];
      b.q[1] = ((const uint4*)bp)[2];
      acc[j] = __builtin_amdgcn_wmma_f32_16x16x32_bf16(
                  false, a.v, false, b.v, (short)0, acc[j], false, false);
    }

#if HAVE_ASYNC_LDS
    if (step + 1 < KSTEPS) __builtin_amdgcn_s_wait_asynccnt(0);
    __syncthreads();                                 // next buffer ready; reads of cur done
#else
    __syncthreads();                                 // reads of (cur^1) from prev step done
    if (step + 1 < KSTEPS) {
      *(uint4*)&Bs[cur ^ 1][bdst0] = nx0;
      *(uint4*)&Bs[cur ^ 1][bdst1] = nx1;
    }
    __syncthreads();
#endif
  }

  #pragma unroll
  for (int j = 0; j < 8; ++j) {
    const int n = nbase + j * 16 + lm;
    #pragma unroll
    for (int r = 0; r < 8; ++r) {
      const int row = m0 + half * 8 + r;             // C/D layout: M = r + 8*half
      if (row < nrows) Y[(size_t)row * NHID + n] = acc[j][r];
    }
  }
}

// ---- SpMM F=256: out[dst] += w * feat[src]; one wave per edge --------------
__global__ __launch_bounds__(256) void k_spmm256(
    const float* __restrict__ feat, const int* __restrict__ src,
    const int* __restrict__ dst, const float* __restrict__ w,
    float* __restrict__ out, int E) {
  int e = (blockIdx.x * blockDim.x + threadIdx.x) >> 5;
  int lane = threadIdx.x & 31;
  if (e >= E) return;
  int s = src[e], d = dst[e];
  float ww = w[e];
  const float4* fp = (const float4*)(feat + (size_t)s * NHID);
  float4 v0 = fp[lane];
  float4 v1 = fp[lane + 32];
  float* op = out + (size_t)d * NHID;
  atomicAdd(op + lane * 4 + 0,       v0.x * ww);
  atomicAdd(op + lane * 4 + 1,       v0.y * ww);
  atomicAdd(op + lane * 4 + 2,       v0.z * ww);
  atomicAdd(op + lane * 4 + 3,       v0.w * ww);
  atomicAdd(op + 128 + lane * 4 + 0, v1.x * ww);
  atomicAdd(op + 128 + lane * 4 + 1, v1.y * ww);
  atomicAdd(op + 128 + lane * 4 + 2, v1.z * ww);
  atomicAdd(op + 128 + lane * 4 + 3, v1.w * ww);
}

// ---- h = relu(acc + b1), packed to bf16 ------------------------------------
__global__ void k_bias_relu_bf16(const float* __restrict__ acc, const float* __restrict__ b1,
                                 uint2* __restrict__ hbf, int total4) {
  int i = blockIdx.x * blockDim.x + threadIdx.x;     // 4 floats per thread
  if (i >= total4) return;
  float4 v = ((const float4*)acc)[i];
  int col = (i * 4) & (NHID - 1);
  v.x = fmaxf(v.x + b1[col + 0], 0.f);
  v.y = fmaxf(v.y + b1[col + 1], 0.f);
  v.z = fmaxf(v.z + b1[col + 2], 0.f);
  v.w = fmaxf(v.w + b1[col + 3], 0.f);
  uint2 o; o.x = pk2(v.x, v.y); o.y = pk2(v.z, v.w);
  hbf[i] = o;
}

// ---- GEMM2: Y[nrows x 64] = H[nrows x 256](bf16) @ W2 (via W2T bf16) -------
__global__ __launch_bounds__(256) void k_gemm_h_w2(
    const unsigned short* __restrict__ H, const unsigned short* __restrict__ W2T,
    float* __restrict__ Y, int nrows) {
  const int lane = threadIdx.x & 31;
  const int wave = threadIdx.x >> 5;
  const int half = lane >> 4;
  const int lm   = lane & 15;
  const int m0   = blockIdx.x * 128 + wave * 16;

  int arow = m0 + lm;
  if (arow >= nrows) arow = nrows - 1;
  const unsigned short* hrow = H + (size_t)arow * NHID;

  v8f acc[4];
  #pragma unroll
  for (int j = 0; j < 4; ++j) acc[j] = (v8f){0,0,0,0,0,0,0,0};

  #pragma unroll
  for (int k0 = 0; k0 < NHID; k0 += 32) {
    const int kA = k0 + half * 8;
    FragU a;
    a.q[0] = *(const uint4*)(hrow + kA);             // K = kA .. kA+7
    a.q[1] = *(const uint4*)(hrow + kA + 16);        // K = kA+16 .. kA+23
    #pragma unroll
    for (int j = 0; j < 4; ++j) {
      const int n = j * 16 + lm;
      const unsigned short* bp = W2T + (size_t)n * NHID + k0 + half * 8;
      FragU b;
      b.q[0] = ((const uint4*)bp)[0];
      b.q[1] = ((const uint4*)bp)[2];
      acc[j] = __builtin_amdgcn_wmma_f32_16x16x32_bf16(
                  false, a.v, false, b.v, (short)0, acc[j], false, false);
    }
  }

  #pragma unroll
  for (int j = 0; j < 4; ++j) {
    const int n = j * 16 + lm;
    #pragma unroll
    for (int r = 0; r < 8; ++r) {
      const int row = m0 + half * 8 + r;
      if (row < nrows) Y[(size_t)row * NCLASS + n] = acc[j][r];
    }
  }
}

// ---- SpMM F=64: one wave per edge, 2 floats per lane -----------------------
__global__ __launch_bounds__(256) void k_spmm64(
    const float* __restrict__ feat, const int* __restrict__ src,
    const int* __restrict__ dst, const float* __restrict__ w,
    float* __restrict__ out, int E) {
  int e = (blockIdx.x * blockDim.x + threadIdx.x) >> 5;
  int lane = threadIdx.x & 31;
  if (e >= E) return;
  int s = src[e], d = dst[e];
  float ww = w[e];
  float2 v = ((const float2*)(feat + (size_t)s * NCLASS))[lane];
  float* op = out + (size_t)d * NCLASS + lane * 2;
  atomicAdd(op + 0, v.x * ww);
  atomicAdd(op + 1, v.y * ww);
}

// ---- row-wise (logits + b2) -> log_softmax, in place; wave per row ---------
__global__ __launch_bounds__(256) void k_logsoftmax(
    float* __restrict__ logits, const float* __restrict__ b2, int nrows) {
  int row  = (blockIdx.x * blockDim.x + threadIdx.x) >> 5;
  int lane = threadIdx.x & 31;
  if (row >= nrows) return;
  float2* rp = (float2*)(logits + (size_t)row * NCLASS);
  float2 v = rp[lane];
  v.x += b2[lane * 2 + 0];
  v.y += b2[lane * 2 + 1];
  float m = fmaxf(v.x, v.y);
  #pragma unroll
  for (int off = 16; off > 0; off >>= 1) m = fmaxf(m, __shfl_xor(m, off, 32));
  float s = __expf(v.x - m) + __expf(v.y - m);
  #pragma unroll
  for (int off = 16; off > 0; off >>= 1) s += __shfl_xor(s, off, 32);
  float l = __logf(s) + m;
  float2 o; o.x = v.x - l; o.y = v.y - l;
  rp[lane] = o;
}

extern "C" void kernel_launch(void* const* d_in, const int* in_sizes, int n_in,
                              void* d_out, int out_size, void* d_ws, size_t ws_size,
                              hipStream_t stream) {
  const float* x    = (const float*)d_in[0];
  const int*   esrc = (const int*)d_in[1];
  const int*   edst = (const int*)d_in[2];
  const float* ew   = (const float*)d_in[3];
  const float* W1   = (const float*)d_in[4];
  const float* b1   = (const float*)d_in[5];
  const float* W2   = (const float*)d_in[6];
  const float* b2   = (const float*)d_in[7];

  const int nnodes = in_sizes[0] / NFEAT;   // 100000
  const int E      = in_sizes[1];           // 3200000

  // workspace layout (~205 MB): [W1T bf16][W2T bf16][bufA][bufB]
  char* ws = (char*)d_ws;
  unsigned short* w1t = (unsigned short*)ws;                       // 256 KB
  unsigned short* w2t = (unsigned short*)(ws + 262144);            // 32 KB
  const size_t szAct  = (size_t)nnodes * NHID * sizeof(float);     // 102.4 MB
  float* xw1  = (float*)(ws + 294912);                             // bufA
  float* acc1 = (float*)(ws + 294912 + szAct);                     // bufB
  unsigned short* hbf = (unsigned short*)xw1;                      // reuse bufA (xw1 dead)
  float* hw2  = acc1;                                              // reuse bufB (acc1 dead)
  float* logits = (float*)d_out;

  // weight convert + transpose to bf16
  k_cvt_wT<<<(NFEAT * NHID + 255) / 256, 256, 0, stream>>>(W1, w1t, NFEAT, NHID);
  k_cvt_wT<<<(NHID * NCLASS + 255) / 256, 256, 0, stream>>>(W2, w2t, NHID, NCLASS);

  // layer 1 GEMM (WMMA bf16, async-LDS double-buffered B tiles)
  dim3 g1((nnodes + 127) / 128, 2);
  k_gemm_x_w1<<<g1, 256, 0, stream>>>(x, w1t, xw1, nnodes);

  // layer 1 SpMM (zero accumulator first; graph-capturable memset)
  (void)hipMemsetAsync(acc1, 0, szAct, stream);
  k_spmm256<<<(E + 7) / 8, 256, 0, stream>>>(xw1, esrc, edst, ew, acc1, E);

  // bias + relu + pack bf16
  const int total4 = nnodes * NHID / 4;
  k_bias_relu_bf16<<<(total4 + 255) / 256, 256, 0, stream>>>(acc1, b1, (uint2*)hbf, total4);

  // layer 2 GEMM (WMMA bf16)
  dim3 g2((nnodes + 127) / 128, 1);
  k_gemm_h_w2<<<g2, 256, 0, stream>>>(hbf, w2t, hw2, nnodes);

  // layer 2 SpMM into d_out, then bias + log_softmax in place
  (void)hipMemsetAsync(logits, 0, (size_t)nnodes * NCLASS * sizeof(float), stream);
  k_spmm64<<<(E + 7) / 8, 256, 0, stream>>>(hw2, esrc, edst, ew, logits, E);
  k_logsoftmax<<<(nnodes * 32 + 255) / 256, 256, 0, stream>>>(logits, b2, nnodes);
}
